// MultiHeadAttention_40587440947332
// MI455X (gfx1250) — matrix operand order repaired and checked by reading, not compiled, not run on hardware
//
#include <hip/hip_runtime.h>

#define BATCH 2
#define SEQ   2048
#define NST   1024
#define NH    16
#define DK    64
#define MROWS (BATCH * SEQ)   // 4096 rows for all projection GEMMs

typedef __attribute__((ext_vector_type(16))) __bf16 v16bf;
typedef __attribute__((ext_vector_type(8)))  float  v8f;

static __device__ __forceinline__ v8f v8f_zero() {
  v8f z = {0.f, 0.f, 0.f, 0.f, 0.f, 0.f, 0.f, 0.f};
  return z;
}

// native v_exp_f32 (base-2)
static __device__ __forceinline__ float fast_exp2(float x) {
  return __builtin_amdgcn_exp2f(x);
}

static __device__ __forceinline__ v8f wmma_bf16(v16bf a, v16bf b, v8f c) {
  // D = A(16x32 bf16) * B(32x16 bf16) + C(16x16 f32)
  return __builtin_amdgcn_wmma_f32_16x16x32_bf16(false, a, false, b,
                                                 (short)0, c, false, false);
}

// A-fragment (16x32, bf16 source). p = row_base + k0 + 16*half.
// elems 0..15 -> K = k0 + 16*half + 0..15
static __device__ __forceinline__ v16bf frag_a_bf16(const __bf16* p) {
  v16bf a;
#pragma unroll
  for (int i = 0; i < 8; ++i) { a[i] = p[i]; a[8 + i] = p[8 + i]; }
  return a;
}

// A-fragment with on-the-fly fp32 -> bf16 convert.
static __device__ __forceinline__ v16bf frag_a_f32(const float* p) {
  v16bf a;
#pragma unroll
  for (int i = 0; i < 8; ++i) {
    a[i]     = (__bf16)p[i];
    a[8 + i] = (__bf16)p[8 + i];
  }
  return a;
}

// B-fragment (32x16) where B[k][n] comes from a K-contiguous row:
// p = row_base(n) + k0 + 16*half ; lane's 16 values are contiguous in K.
static __device__ __forceinline__ v16bf frag_b_kmajor(const __bf16* p) {
  v16bf b;
#pragma unroll
  for (int i = 0; i < 16; ++i) b[i] = p[i];
  return b;
}

// ---------------------------------------------------------------------------
// fp32 -> bf16 conversion (weights only), 4 elements/thread
// ---------------------------------------------------------------------------
__global__ void f32_to_bf16(const float* __restrict__ src,
                            __bf16* __restrict__ dst, int n) {
  int i = (blockIdx.x * blockDim.x + threadIdx.x) * 4;
  if (i + 3 < n) {
#pragma unroll
    for (int j = 0; j < 4; ++j) dst[i + j] = (__bf16)src[i + j];
  }
}

// ---------------------------------------------------------------------------
// Y[M,N] = A[M,K] @ W[N,K]^T + bias[N]
// One wave -> 16x64 tile (4 WMMA accumulators sharing one A-fragment).
// ---------------------------------------------------------------------------
template <bool A_IS_F32, bool OUT_F32>
__global__ void __launch_bounds__(256)
gemm_xwt(const void* __restrict__ Av, const __bf16* __restrict__ W,
         const float* __restrict__ bias, void* __restrict__ Yv,
         int M, int N, int K) {
  const int lane = threadIdx.x & 31;
  const int l    = lane & 15;
  const int half = lane >> 4;
  const int wave = (blockIdx.x * blockDim.x + threadIdx.x) >> 5;
  const int ntg  = N >> 6;               // number of 64-wide column groups
  const int m0   = (wave / ntg) << 4;
  const int n0   = (wave % ntg) << 6;

  v8f c[4];
#pragma unroll
  for (int j = 0; j < 4; ++j) c[j] = v8f_zero();

  for (int k0 = 0; k0 < K; k0 += 32) {
    v16bf a;
    if constexpr (A_IS_F32) {
      const float* pa = (const float*)Av + (size_t)(m0 + l) * K + k0 + 16 * half;
      __builtin_prefetch(pa + 32, 0, 1);
      a = frag_a_f32(pa);
    } else {
      const __bf16* pa = (const __bf16*)Av + (size_t)(m0 + l) * K + k0 + 16 * half;
      __builtin_prefetch(pa + 32, 0, 1);
      a = frag_a_bf16(pa);
    }
#pragma unroll
    for (int j = 0; j < 4; ++j) {
      v16bf b = frag_b_kmajor(W + (size_t)(n0 + j * 16 + l) * K + k0 + 16 * half);
      c[j] = wmma_bf16(a, b, c[j]);
    }
  }

#pragma unroll
  for (int j = 0; j < 4; ++j) {
    const int col = n0 + j * 16 + l;
    const float bj = bias[col];
#pragma unroll
    for (int r = 0; r < 8; ++r) {
      const int row = m0 + r + 8 * half;
      const float val = c[j][r] + bj;
      if constexpr (OUT_F32)
        ((float*)Yv)[(size_t)row * N + col] = val;
      else
        ((__bf16*)Yv)[(size_t)row * N + col] = (__bf16)val;
    }
  }
}

// ---------------------------------------------------------------------------
// Cooperative flash attention, transposed formulation.
// One block = 8 waves = 128 queries of one (b,h). Per 32-key chunk the block
// stages K (row-major) and V (transposed) into LDS once, shared by all waves.
// Each wave computes S^T = K(32x64) x Q^T(64x16): queries are C-fragment
// COLUMNS -> softmax stats are per-lane (in-register reduce + one shfl_xor 16),
// then O^T += V^T(64x32) x P^T(32x16).
// qh/kh/vh/att are [B*S, 1024] bf16; head h occupies columns h*64..h*64+63.
// ---------------------------------------------------------------------------
__global__ void __launch_bounds__(256)
attn_fwd(const __bf16* __restrict__ qh, const __bf16* __restrict__ kh,
         const __bf16* __restrict__ vh, __bf16* __restrict__ att) {
  constexpr int LDK = 72;  // Ks pitch: 16-lane frag bases hit distinct banks
  constexpr int LDV = 40;  // Vt pitch: same property
  constexpr int LDP = 40;  // P^T bounce pitch
  __shared__ __bf16 Ks[32 * LDK];      // 32 keys x 64 d   (row-major)
  __shared__ __bf16 Vt[64 * LDV];      // 64 d   x 32 keys (transposed)
  __shared__ __bf16 Pl[8][16 * LDP];   // per-wave P^T bounce: [query][key]

  const int tid  = threadIdx.x;
  const int lane = tid & 31;
  const int l    = lane & 15;
  const int half = lane >> 4;
  const int wv   = tid >> 5;

  const int bh = blockIdx.x >> 4;            // 0..31
  const int b  = bh >> 4;
  const int h  = bh & (NH - 1);
  const int qw = (blockIdx.x & 15) * 128 + wv * 16;   // this wave's 16 queries

  const size_t base = (size_t)b * SEQ * NST + (size_t)h * DK;
  const __bf16* Q  = qh + base;
  const __bf16* Kp = kh + base;
  const __bf16* Vp = vh + base;

  // Q^T B-fragments: B[k=d][n=query], K-contiguous per lane. Register-resident.
  v16bf qb[2];
#pragma unroll
  for (int kk = 0; kk < 2; ++kk)
    qb[kk] = frag_b_kmajor(Q + (size_t)(qw + l) * NST + kk * 32 + 16 * half);

  v8f o[4];                      // O^T tiles: row d = n*16 + r + 8*half, col q = l
#pragma unroll
  for (int n = 0; n < 4; ++n) o[n] = v8f_zero();
  float mq = -1e30f, lq = 0.f;   // per-query stats (query = lane l, both halves)

  // softmax in exp2 domain: fold 1/sqrt(64) * log2(e) into one scale
  const float SCL = 0.125f * 1.44269504088896f;

  const int srow = tid >> 3;        // staging: key row 0..31
  const int scol = (tid & 7) * 8;   // staging: d col 0,8,..,56

  for (int key0 = 0; key0 < SEQ; key0 += 32) {
    __syncthreads();               // previous chunk's LDS reads done
    {
      const __bf16* kg = Kp + (size_t)(key0 + srow) * NST + scol;
      const __bf16* vg = Vp + (size_t)(key0 + srow) * NST + scol;
      __builtin_prefetch(kg + 32 * NST, 0, 1);
      __builtin_prefetch(vg + 32 * NST, 0, 1);
#pragma unroll
      for (int i = 0; i < 8; ++i) Ks[srow * LDK + scol + i] = kg[i];
#pragma unroll
      for (int i = 0; i < 8; ++i) Vt[(scol + i) * LDV + srow] = vg[i];
    }
    __syncthreads();               // staging visible to all waves

    // ---- S^T[32 keys x 16 q]: two key sub-tiles, K-dim = 64 (2 wmma each)
    v8f st[2] = {v8f_zero(), v8f_zero()};
#pragma unroll
    for (int j = 0; j < 2; ++j)
#pragma unroll
      for (int kk = 0; kk < 2; ++kk) {
        v16bf ak = frag_a_bf16(Ks + (j * 16 + l) * LDK + kk * 32 + 16 * half);
        st[j] = wmma_bf16(ak, qb[kk], st[j]);
      }

    // ---- per-query online softmax: in-lane reduce + ONE cross-half shuffle
    float tmax = -1e30f;
#pragma unroll
    for (int j = 0; j < 2; ++j)
#pragma unroll
      for (int r = 0; r < 8; ++r) {
        st[j][r] *= SCL;
        tmax = fmaxf(tmax, st[j][r]);
      }
    tmax = fmaxf(tmax, __shfl_xor(tmax, 16, 32));
    const float mnew  = fmaxf(mq, tmax);
    const float alpha = fast_exp2(mq - mnew);
    mq = mnew;

    float psum = 0.f;
#pragma unroll
    for (int j = 0; j < 2; ++j)
#pragma unroll
      for (int r = 0; r < 8; ++r) {
        const float p = fast_exp2(st[j][r] - mnew);
        Pl[wv][l * LDP + j * 16 + r + 8 * half] = (__bf16)p;  // P^T -> [q][key]
        psum += p;
      }
    psum += __shfl_xor(psum, 16, 32);
    lq = lq * alpha + psum;
#pragma unroll
    for (int n = 0; n < 4; ++n)
#pragma unroll
      for (int r = 0; r < 8; ++r) o[n][r] *= alpha;
    asm volatile("s_wait_dscnt 0" ::: "memory");  // same-wave DS RAW fence

    // ---- O^T += V^T(64x32) x P^T(32x16)
    v16bf pb = frag_b_kmajor(&Pl[wv][l * LDP + 16 * half]);
#pragma unroll
    for (int n = 0; n < 4; ++n) {
      v16bf av = frag_a_bf16(Vt + (n * 16 + l) * LDV + 16 * half);
      o[n] = wmma_bf16(av, pb, o[n]);
    }
  }

  // ---- normalize (per-lane scalar) and store: 16B contiguous per lane/tile
  const float inv = 1.0f / lq;
#pragma unroll
  for (int n = 0; n < 4; ++n)
#pragma unroll
    for (int r = 0; r < 8; ++r)
      att[((size_t)b * SEQ + qw + l) * NST + h * DK + n * 16 + 8 * half + r] =
          (__bf16)(o[n][r] * inv);
}

// ---------------------------------------------------------------------------
extern "C" void kernel_launch(void* const* d_in, const int* in_sizes, int n_in,
                              void* d_out, int out_size, void* d_ws, size_t ws_size,
                              hipStream_t stream) {
  const float* q  = (const float*)d_in[0];
  const float* k  = (const float*)d_in[1];
  const float* v  = (const float*)d_in[2];
  const float* Wq = (const float*)d_in[3];
  const float* bq = (const float*)d_in[4];
  const float* Wk = (const float*)d_in[5];
  const float* bk = (const float*)d_in[6];
  const float* Wv = (const float*)d_in[7];
  const float* bv = (const float*)d_in[8];
  const float* Wo = (const float*)d_in[9];
  const float* bo = (const float*)d_in[10];

  char* ws = (char*)d_ws;
  const size_t WB = (size_t)NST * NST * sizeof(__bf16);    // 2 MB per weight
  const size_t HB = (size_t)MROWS * NST * sizeof(__bf16);  // 8 MB per activation
  __bf16* Wq_bf = (__bf16*)(ws + 0 * WB);
  __bf16* Wk_bf = (__bf16*)(ws + 1 * WB);
  __bf16* Wv_bf = (__bf16*)(ws + 2 * WB);
  __bf16* Wo_bf = (__bf16*)(ws + 3 * WB);
  __bf16* qh    = (__bf16*)(ws + 4 * WB);
  __bf16* kh    = (__bf16*)(ws + 4 * WB + 1 * HB);
  __bf16* vh    = (__bf16*)(ws + 4 * WB + 2 * HB);
  __bf16* att   = (__bf16*)(ws + 4 * WB + 3 * HB);   // total 40 MB of ws

  // 1) weights fp32 -> bf16
  const int wn = NST * NST;                      // 1M elements each
  f32_to_bf16<<<wn / 1024, 256, 0, stream>>>(Wq, Wq_bf, wn);
  f32_to_bf16<<<wn / 1024, 256, 0, stream>>>(Wk, Wk_bf, wn);
  f32_to_bf16<<<wn / 1024, 256, 0, stream>>>(Wv, Wv_bf, wn);
  f32_to_bf16<<<wn / 1024, 256, 0, stream>>>(Wo, Wo_bf, wn);

  // 2) Q/K/V projections: (4096/16)*(1024/64) = 4096 waves -> 512 blocks
  gemm_xwt<true, false><<<512, 256, 0, stream>>>(q, Wq_bf, bq, qh, MROWS, NST, NST);
  gemm_xwt<true, false><<<512, 256, 0, stream>>>(k, Wk_bf, bk, kh, MROWS, NST, NST);
  gemm_xwt<true, false><<<512, 256, 0, stream>>>(v, Wv_bf, bv, vh, MROWS, NST, NST);

  // 3) cooperative flash attention: B*H*16 = 512 blocks x 256 threads
  attn_fwd<<<512, 256, 0, stream>>>(qh, kh, vh, att);

  // 4) output projection, fp32 result straight into d_out
  gemm_xwt<false, true><<<512, 256, 0, stream>>>(att, Wo_bf, bo, d_out, MROWS, NST, NST);
}
